// HardNegativeMiner_59734405153458
// MI455X (gfx1250) — matrix operand order repaired and checked by reading, not compiled, not run on hardware
//
#include <hip/hip_runtime.h>
#include <hip/hip_bf16.h>

// ---------------------------------------------------------------------------
// HardNegativeMiner for MI455X (gfx1250, wave32)
//   Pass 1: row-L2-normalize z (fp32) -> Zn (bf16, 8 MB workspace; stays
//           resident in the 192 MB L2 for pass 2)
//   Pass 2: fused  S = Zn[src] @ Zn^T  via v_wmma_f32_16x16x32_bf16
//           + Gumbel-max categorical sampling streamed in the epilogue
//           + final alpha-mix of the selected rows.
// Software-pipelined: B-fragment loads for tile(it) are issued as one clause,
// then the Gumbel epilogue of tile(it-1) executes while they are in flight,
// then the WMMA chain consumes them. Two accumulators break the serial
// WMMA->WMMA RAW chain.
// ---------------------------------------------------------------------------

typedef __attribute__((ext_vector_type(16))) __bf16 v16bf;
typedef __attribute__((ext_vector_type(8)))  float  v8f;

union Frag {
    v16bf v;
    uint4 q[2];
};

#define D         256           // feature dim
#define D_U32     128           // row stride of Zn in uints (256 bf16)
#define NEG_INF   (-3.4e38f)

__device__ __forceinline__ unsigned hash32(unsigned x) {
    x ^= x >> 16; x *= 0x7feb352dU;
    x ^= x >> 15; x *= 0x846ca68bU;
    x ^= x >> 16;
    return x;
}
__device__ __forceinline__ float u01(unsigned h) {
    return ((float)h + 0.5f) * 2.3283064365386963e-10f;   // (h+0.5) * 2^-32
}
__device__ __forceinline__ unsigned f2bf_rne(float f) {   // fp32 -> bf16 RNE
    unsigned u = __float_as_uint(f);
    return (u + 0x7FFFu + ((u >> 16) & 1u)) >> 16;
}

// ---------------------------------------------------------------------------
// Kernel 1: normalize rows of z, store bf16 rows into Zn.
// One wave per row; lane handles 8 contiguous floats (coalesced 1KB/wave).
// ---------------------------------------------------------------------------
__global__ __launch_bounds__(256)
void hnm_normalize(const float* __restrict__ z, unsigned* __restrict__ zn, int B) {
    const int wave = threadIdx.x >> 5;
    const int lane = threadIdx.x & 31;
    const int row  = blockIdx.x * 8 + wave;
    if (row >= B) return;                       // wave-uniform exit

    const float4* p = (const float4*)(z + row * D + lane * 8);
    float4 a = p[0];
    float4 b = p[1];
    float ss = a.x*a.x + a.y*a.y + a.z*a.z + a.w*a.w
             + b.x*b.x + b.y*b.y + b.z*b.z + b.w*b.w;
#pragma unroll
    for (int off = 16; off > 0; off >>= 1)
        ss += __shfl_xor(ss, off, 32);

    const float inv = 1.0f / fmaxf(sqrtf(ss), 1e-12f);    // F.normalize eps clamp

    unsigned q0 = f2bf_rne(a.x*inv) | (f2bf_rne(a.y*inv) << 16);
    unsigned q1 = f2bf_rne(a.z*inv) | (f2bf_rne(a.w*inv) << 16);
    unsigned q2 = f2bf_rne(b.x*inv) | (f2bf_rne(b.y*inv) << 16);
    unsigned q3 = f2bf_rne(b.z*inv) | (f2bf_rne(b.w*inv) << 16);
    *(uint4*)(zn + row * D_U32 + lane * 4) = make_uint4(q0, q1, q2, q3);
}

// ---------------------------------------------------------------------------
// Kernel 2: per block -> 16 sampled rows.
// ---------------------------------------------------------------------------
__global__ __launch_bounds__(256)
void hnm_mine(const float* __restrict__ z, const unsigned* __restrict__ zn,
              float* __restrict__ out, int B, int n_hard) {
    __shared__ unsigned short ldsA[16 * D];    // 8 KB A tile (bf16)
    __shared__ float sBest[8 * 16];
    __shared__ int   sIdx [8 * 16];
    __shared__ int   sSrc[16];
    __shared__ int   sTgt[16];

    const int tid  = threadIdx.x;
    const int lane = tid & 31;
    const int wave = tid >> 5;
    const int base = blockIdx.x * 16;          // first sample index of block

    if (tid < 16)
        sSrc[tid] = (int)(hash32(0x9E3779B9u ^ (unsigned)(base + tid)) & (unsigned)(B - 1));
    __syncthreads();

    // --- stage A tile: 16 rows x 512B, 32B per thread -----------------------
    {
        const int r = tid >> 4;                // 0..15  : which sampled row
        const int c = (tid & 15) * 8;          // uint offset inside row
        const uint4* sp = (const uint4*)(zn + sSrc[r] * D_U32 + c);
        uint4 x0 = sp[0], x1 = sp[1];
        *(uint4*)&ldsA[r * D + c * 2]     = x0;
        *(uint4*)&ldsA[r * D + c * 2 + 8] = x1;
    }
    __syncthreads();

    // --- preload the 8 A fragments (K chunks of 32) into registers ----------
    // A 16x32 bf16 layout: lanes 0-15 (M=l) hold K 0-7 & 16-23,
    //                      lanes 16-31 hold K 8-15 & 24-31.
    Frag A[8];
    {
        const int m    = lane & 15;
        const int half = lane >> 4;
#pragma unroll
        for (int kk = 0; kk < 8; ++kk) {
            A[kk].q[0] = *(const uint4*)&ldsA[m * D + kk * 32 + half * 8];
            A[kk].q[1] = *(const uint4*)&ldsA[m * D + kk * 32 + 16 + half * 8];
        }
    }

    const int mrow   = (lane >> 4) << 3;       // C rows: 0 (lanes 0-15) or 8
    const int ncol   = lane & 15;              // C col within tile
    const int myHalf = lane >> 4;

    // per-slot loop-invariants (avoid LDS reads / v_mul_lo in the hot loop)
    float    best[8];
    int      bidx[8];
    int      srcOf[8];
    unsigned gb[8];
#pragma unroll
    for (int r = 0; r < 8; ++r) {
        const int m = r + mrow;
        best[r]  = NEG_INF;
        bidx[r]  = 0;
        srcOf[r] = sSrc[m];
        gb[r]    = (unsigned)((base + m) * B);
    }

    v8f c0 = {}, c1 = {};
    int pn0 = -1;                              // previous tile's column base

    const int nIter = B >> 7;                  // B / (8 waves * 16 cols) = 128
    for (int it = 0; it < nIter; ++it) {
        const int n0 = ((it * 8 + wave) << 4); // this wave's 16-column tile

        // ---- issue ALL 16 B-fragment loads for this tile (one clause) ------
        const unsigned* brow = zn + (n0 + ncol) * D_U32;   // column j = row j of Zn
        uint4 bq[16];
#pragma unroll
        for (int kk = 0; kk < 8; ++kk) {
            bq[2*kk]   = *(const uint4*)(brow + kk * 16 + myHalf * 4);
            bq[2*kk+1] = *(const uint4*)(brow + kk * 16 + 8 + myHalf * 4);
        }

        // ---- Gumbel-max epilogue of the PREVIOUS tile (hides load latency) -
        if (pn0 >= 0) {
            const int j = pn0 + ncol;
#pragma unroll
            for (int r = 0; r < 8; ++r) {
                float logit = (c0[r] + c1[r]) * 10.0f;       // sim / 0.1
                if (j == srcOf[r]) logit = NEG_INF;          // exclude diagonal
                const unsigned h = hash32(0x85EBCA6Bu ^ (gb[r] + (unsigned)j));
                const float gum = -__logf(fmaxf(-__logf(u01(h)), 1e-30f));
                const float s = logit + gum;
                if (s > best[r]) { best[r] = s; bidx[r] = j; }
            }
        }

        // ---- WMMA chain: two accumulators break the serial RAW chain -------
        c0 = (v8f){}; c1 = (v8f){};
#pragma unroll
        for (int kk = 0; kk < 8; kk += 2) {
            Frag f0; f0.q[0] = bq[2*kk];     f0.q[1] = bq[2*kk+1];
            c0 = __builtin_amdgcn_wmma_f32_16x16x32_bf16(
                     false, A[kk].v,   false, f0.v, (short)0, c0, false, false);
            Frag f1; f1.q[0] = bq[2*kk+2];   f1.q[1] = bq[2*kk+3];
            c1 = __builtin_amdgcn_wmma_f32_16x16x32_bf16(
                     false, A[kk+1].v, false, f1.v, (short)0, c1, false, false);
        }
        pn0 = n0;
    }

    // ---- drain the pipeline: epilogue of the last tile ----------------------
    {
        const int j = pn0 + ncol;
#pragma unroll
        for (int r = 0; r < 8; ++r) {
            float logit = (c0[r] + c1[r]) * 10.0f;
            if (j == srcOf[r]) logit = NEG_INF;
            const unsigned h = hash32(0x85EBCA6Bu ^ (gb[r] + (unsigned)j));
            const float gum = -__logf(fmaxf(-__logf(u01(h)), 1e-30f));
            const float s = logit + gum;
            if (s > best[r]) { best[r] = s; bidx[r] = j; }
        }
    }

    // --- reduce across the 16 lanes that share each C row --------------------
#pragma unroll
    for (int r = 0; r < 8; ++r) {
#pragma unroll
        for (int off = 8; off > 0; off >>= 1) {
            const float os = __shfl_xor(best[r], off, 16);
            const int   oi = __shfl_xor(bidx[r], off, 16);
            if (os > best[r] || (os == best[r] && oi < bidx[r])) {
                best[r] = os; bidx[r] = oi;
            }
        }
    }
    if ((lane & 15) == 0) {
#pragma unroll
        for (int r = 0; r < 8; ++r) {
            sBest[wave * 16 + r + mrow] = best[r];
            sIdx [wave * 16 + r + mrow] = bidx[r];
        }
    }
    __syncthreads();

    // --- reduce across the 8 waves ------------------------------------------
    if (tid < 16) {
        float bb = sBest[tid];
        int   bi = sIdx[tid];
        for (int w = 1; w < 8; ++w) {
            const float os = sBest[w * 16 + tid];
            const int   oi = sIdx [w * 16 + tid];
            if (os > bb || (os == bb && oi < bi)) { bb = os; bi = oi; }
        }
        sTgt[tid] = bi;
    }
    __syncthreads();

    // --- mix: out[g] = alpha*z[src] + (1-alpha)*z[tgt], col = tid ------------
    for (int r = 0; r < 16; ++r) {
        const int g = base + r;
        if (g >= n_hard) continue;
        const float alpha = u01(hash32(0xC2B2AE35u ^ (unsigned)g)) * 0.5f; // * MIX_ALPHA
        const float zs = z[sSrc[r] * D + tid];
        const float zt = z[sTgt[r] * D + tid];
        out[g * D + tid] = alpha * zs + (1.0f - alpha) * zt;
    }
}

// ---------------------------------------------------------------------------
extern "C" void kernel_launch(void* const* d_in, const int* in_sizes, int n_in,
                              void* d_out, int out_size, void* d_ws, size_t ws_size,
                              hipStream_t stream) {
    const float* z = (const float*)d_in[0];
    const int B = in_sizes[0] / D;             // 16384
    int n_hard = B / 5;                        // int(B * 0.2) for B = 16384
    if (n_hard < 1) n_hard = 1;

    unsigned* zn = (unsigned*)d_ws;            // bf16 normalized z: B*256*2 bytes

    hnm_normalize<<<(B + 7) / 8, 256, 0, stream>>>(z, zn, B);

    const int nblocks = (n_hard + 15) / 16;
    hnm_mine<<<nblocks, 256, 0, stream>>>(z, zn, (float*)d_out, B, n_hard);
}
